// DiTBlockWithAdaLNZero_86947317940756
// MI455X (gfx1250) — matrix operand order repaired and checked by reading, not compiled, not run on hardware
//
#include <hip/hip_runtime.h>

// ---------------------------------------------------------------------------
// DiT block with AdaLN-Zero for MI455X (gfx1250, wave32).
// GEMMs: v_wmma_f32_16x16x32_bf16, 128x128 block tile staged through LDS
// (ds_load_b128 fragment feeds), 32x64 per-wave C tile, f32 accumulate.
// Attention (window=16, ~1 GFLOP) via scalar VALU online softmax.
// ---------------------------------------------------------------------------

typedef __attribute__((ext_vector_type(16))) __bf16 v16bf;
typedef __attribute__((ext_vector_type(8)))  __bf16 v8bf;
typedef __attribute__((ext_vector_type(8)))  float  v8f;

#define S_    2048
#define B_    4
#define D_    1024
#define H_    16
#define DH_   64
#define MLPH_ 4096
#define WIN_  16
#define SIXD_ 6144
#define NTOK_ 8192   // S_*B_, token t = s*B_ + b  (matches [S,B,D] contiguity)

#define MODE_QKV  0
#define MODE_GATE 1
#define MODE_RELU 2

#define KBLK  64
#define LDL   72     // padded LDS row stride (elements): 144 B, 16B-aligned rows

// -------------------------- f32 -> bf16 weight convert ---------------------
__global__ void cvt_bf16_kernel(const float* __restrict__ src,
                                __bf16* __restrict__ dst, int n) {
  int i = blockIdx.x * blockDim.x + threadIdx.x;
  int stride = gridDim.x * blockDim.x;
  for (; i < n; i += stride) dst[i] = (__bf16)src[i];
}

// -------------------------- adaLN: mod = relu(c) @ ada_w^T + ada_b ---------
__global__ void ada_mod_kernel(const float* __restrict__ c,
                               const float* __restrict__ ada_w,
                               const float* __restrict__ ada_b,
                               float* __restrict__ mod) {
  int gid = blockIdx.x * blockDim.x + threadIdx.x;
  if (gid >= B_ * SIXD_) return;
  int o = gid % SIXD_;
  int b = gid / SIXD_;
  const float4* cw = (const float4*)(c + (size_t)b * D_);
  const float4* ww = (const float4*)(ada_w + (size_t)o * D_);
  float acc = 0.f;
  for (int k = 0; k < D_ / 4; ++k) {
    float4 cv = cw[k], wv = ww[k];
    acc = fmaf(fmaxf(cv.x, 0.f), wv.x, acc);
    acc = fmaf(fmaxf(cv.y, 0.f), wv.y, acc);
    acc = fmaf(fmaxf(cv.z, 0.f), wv.z, acc);
    acc = fmaf(fmaxf(cv.w, 0.f), wv.w, acc);
  }
  mod[(size_t)b * SIXD_ + o] = acc + ada_b[o];
}

// -------------------------- LayerNorm + AdaLN modulation -> bf16 -----------
// xm = (x - mean)/sqrt(var+eps) * (1 + sc[b,:]) + sh[b,:]
__global__ void ln_mod_kernel(const float* __restrict__ x,
                              const float* __restrict__ mod,
                              __bf16* __restrict__ xm,
                              int shOff, int scOff) {
  __shared__ float r1[256], r2[256];
  int t   = blockIdx.x;          // token
  int b   = t & (B_ - 1);
  int tid = threadIdx.x;         // 256 threads, 4 elems each
  float4 v = *(const float4*)(x + (size_t)t * D_ + tid * 4);
  r1[tid] = v.x + v.y + v.z + v.w;
  r2[tid] = v.x * v.x + v.y * v.y + v.z * v.z + v.w * v.w;
  __syncthreads();
  for (int o = 128; o; o >>= 1) {
    if (tid < o) { r1[tid] += r1[tid + o]; r2[tid] += r2[tid + o]; }
    __syncthreads();
  }
  float mean = r1[0] * (1.f / D_);
  float var  = r2[0] * (1.f / D_) - mean * mean;
  float rstd = rsqrtf(var + 1e-6f);
  const float* sc = mod + (size_t)b * SIXD_ + scOff + tid * 4;
  const float* sh = mod + (size_t)b * SIXD_ + shOff + tid * 4;
  float xv[4] = {v.x, v.y, v.z, v.w};
  __bf16* op = xm + (size_t)t * D_ + tid * 4;
  #pragma unroll
  for (int i = 0; i < 4; ++i) {
    float y = (xv[i] - mean) * rstd * (1.f + sc[i]) + sh[i];
    op[i] = (__bf16)y;
  }
}

// -------------------------- bf16 WMMA GEMM: C = A @ W^T + bias -------------
// A: [NTOK_, K] bf16 row-major.  W: [N, K] bf16 row-major (== B col-major).
// Block tile 128(M) x 128(N), KBLK=64, A/B panels staged in LDS.
// 8 waves = 4(M) x 2(N); each wave computes a 32x64 C tile (8 accumulators).
__global__ __launch_bounds__(256, 2)
void gemm_wmma_kernel(const __bf16* __restrict__ A,
                      const __bf16* __restrict__ W,
                      const float*  __restrict__ bias,
                      int K, int N, int mode, int gateOff,
                      const float*  __restrict__ mod,
                      const float*  __restrict__ residual,
                      float* __restrict__ outF,
                      __bf16* __restrict__ outB,
                      float* __restrict__ qkv) {
  __shared__ alignas(16) __bf16 As[128 * LDL];   // 18 KB
  __shared__ alignas(16) __bf16 Bs[128 * LDL];   // 18 KB

  int tid   = threadIdx.x;
  int lane  = tid & 31;
  int wave  = tid >> 5;
  int l15   = lane & 15;
  int hi    = lane >> 4;        // K-half select per ISA A/B fragment layouts
  int wm    = (wave & 3) * 32;  // wave's M offset inside block tile
  int wn    = (wave >> 2) * 64; // wave's N offset inside block tile
  int mBlk  = blockIdx.x * 128;
  int nBlk  = blockIdx.y * 128;

  v8f acc[2][4];
  #pragma unroll
  for (int r = 0; r < 2; ++r)
    #pragma unroll
    for (int f = 0; f < 4; ++f)
      acc[r][f] = v8f{0.f, 0.f, 0.f, 0.f, 0.f, 0.f, 0.f, 0.f};

  for (int kb = 0; kb < K; kb += KBLK) {
    // Cooperative stage: 128 rows x 64 K elems = 1024 16-byte chunks each
    // for A and B; 4 chunks per thread per panel.
    #pragma unroll
    for (int i = 0; i < 4; ++i) {
      int idx = i * 256 + tid;
      int row = idx >> 3;           // 0..127
      int ch  = (idx & 7) * 8;      // element offset 0..56
      const __bf16* pa = A + (size_t)(mBlk + row) * K + kb + ch;
      const __bf16* pb = W + (size_t)(nBlk + row) * K + kb + ch;
      *(v8bf*)&As[row * LDL + ch] = *(const v8bf*)pa;
      *(v8bf*)&Bs[row * LDL + ch] = *(const v8bf*)pb;
      if (kb + KBLK < K) {          // global_prefetch_b8 for next K-block
        __builtin_prefetch(pa + KBLK, 0, 0);
        __builtin_prefetch(pb + KBLK, 0, 0);
      }
    }
    __syncthreads();

    #pragma unroll
    for (int ks = 0; ks < KBLK; ks += 32) {
      // Hoist ALL fragment loads of this k-step before the WMMAs so the
      // ds_load_b128s issue as one clause (single dscnt wait ladder)
      // instead of a full s_wait_dscnt 0 drain per WMMA pair.
      v16bf afr[2];
      v16bf bfr[4];
      #pragma unroll
      for (int r = 0; r < 2; ++r) {
        // A 16x32 frag: lanes<16 K=[ks,ks+8)+[ks+16,ks+24); hi lanes +8
        const __bf16* pa = &As[(wm + r * 16 + l15) * LDL + ks + hi * 8];
        v8bf alo = *(const v8bf*)pa;
        v8bf ahi = *(const v8bf*)(pa + 16);
        afr[r] = __builtin_shufflevector(alo, ahi,
            0, 1, 2, 3, 4, 5, 6, 7, 8, 9, 10, 11, 12, 13, 14, 15);
      }
      #pragma unroll
      for (int f = 0; f < 4; ++f) {
        // B 32x16 frag: lanes<16 K=[ks,ks+8)+[ks+16,ks+24); hi lanes +8
        const __bf16* pb = &Bs[(wn + f * 16 + l15) * LDL + ks + hi * 8];
        v8bf blo = *(const v8bf*)pb;
        v8bf bhi = *(const v8bf*)(pb + 16);
        bfr[f] = __builtin_shufflevector(blo, bhi,
            0, 1, 2, 3, 4, 5, 6, 7, 8, 9, 10, 11, 12, 13, 14, 15);
      }
      #pragma unroll
      for (int f = 0; f < 4; ++f) {
        acc[0][f] = __builtin_amdgcn_wmma_f32_16x16x32_bf16(
            false, afr[0], false, bfr[f], (short)0, acc[0][f], false, false);
        acc[1][f] = __builtin_amdgcn_wmma_f32_16x16x32_bf16(
            false, afr[1], false, bfr[f], (short)0, acc[1][f], false, false);
      }
    }
    __syncthreads();
  }

  // Epilogue: C VGPR v, lane L -> M = v + 8*(L>=16), N = L%16 (ISA 7.12.2)
  #pragma unroll
  for (int r = 0; r < 2; ++r) {
    #pragma unroll
    for (int f = 0; f < 4; ++f) {
      int n = nBlk + wn + f * 16 + l15;
      float bv = bias[n];
      #pragma unroll
      for (int v = 0; v < 8; ++v) {
        int m = mBlk + wm + r * 16 + v + hi * 8;   // token t = s*B_ + b
        float val = acc[r][f][v] + bv;
        if (mode == MODE_QKV) {
          int which = n >> 10;            // 0=q 1=k 2=v
          int d = n & (D_ - 1);
          int hh = d >> 6;
          int e  = d & 63;
          int s  = m >> 2;
          int b  = m & 3;
          float scl = (which == 0) ? 0.125f : 1.0f;  // q /= sqrt(64)
          qkv[((((size_t)which * B_ + b) * H_ + hh) * S_ + s) * DH_ + e] = val * scl;
        } else if (mode == MODE_GATE) {
          int b = m & 3;
          float g = mod[(size_t)b * SIXD_ + gateOff + n];
          outF[(size_t)m * D_ + n] = residual[(size_t)m * D_ + n] + g * val;
        } else {  // MODE_RELU -> bf16 hidden
          outB[(size_t)m * N + n] = (__bf16)fmaxf(val, 0.f);
        }
      }
    }
  }
}

// -------------------------- windowed attention (band |i-j|<=16) ------------
// One wave per (b,h,s); dh=64 -> 2 elements per lane; online softmax.
__global__ __launch_bounds__(256)
void attn_window_kernel(const float* __restrict__ qkv,
                        __bf16* __restrict__ out) {
  int wid  = blockIdx.x * 8 + (threadIdx.x >> 5);
  int lane = threadIdx.x & 31;
  int s = wid & (S_ - 1);
  int h = (wid >> 11) & (H_ - 1);
  int b = wid >> 15;
  const float* qb = qkv + (((size_t)0 * B_ + b) * H_ + h) * (size_t)S_ * DH_;
  const float* kb = qkv + (((size_t)1 * B_ + b) * H_ + h) * (size_t)S_ * DH_;
  const float* vb = qkv + (((size_t)2 * B_ + b) * H_ + h) * (size_t)S_ * DH_;
  float q0 = qb[(size_t)s * DH_ + lane];
  float q1 = qb[(size_t)s * DH_ + lane + 32];
  int lo = s - WIN_; if (lo < 0) lo = 0;
  int hiT = s + WIN_; if (hiT > S_ - 1) hiT = S_ - 1;
  float mx = -3.4e38f, l = 0.f, o0 = 0.f, o1 = 0.f;
  for (int t = lo; t <= hiT; ++t) {
    float p = q0 * kb[(size_t)t * DH_ + lane] + q1 * kb[(size_t)t * DH_ + lane + 32];
    #pragma unroll
    for (int off = 16; off; off >>= 1) p += __shfl_xor(p, off, 32);
    float nm = fmaxf(mx, p);
    float es = __expf(mx - nm);   // first iter: exp(-inf)=0
    float ep = __expf(p - nm);
    l  = l * es + ep;
    o0 = o0 * es + ep * vb[(size_t)t * DH_ + lane];
    o1 = o1 * es + ep * vb[(size_t)t * DH_ + lane + 32];
    mx = nm;
  }
  float inv = 1.f / l;
  size_t tok = (size_t)(s * B_ + b) * D_ + h * DH_;
  out[tok + lane]      = (__bf16)(o0 * inv);
  out[tok + lane + 32] = (__bf16)(o1 * inv);
}

// ---------------------------------------------------------------------------
extern "C" void kernel_launch(void* const* d_in, const int* in_sizes, int n_in,
                              void* d_out, int out_size, void* d_ws, size_t ws_size,
                              hipStream_t stream) {
  const float* x          = (const float*)d_in[0];
  const float* c          = (const float*)d_in[1];
  const float* ada_w      = (const float*)d_in[2];
  const float* ada_b      = (const float*)d_in[3];
  const float* in_proj_w  = (const float*)d_in[4];
  const float* in_proj_b  = (const float*)d_in[5];
  const float* out_proj_w = (const float*)d_in[6];
  const float* out_proj_b = (const float*)d_in[7];
  const float* mlp_w1     = (const float*)d_in[8];
  const float* mlp_b1     = (const float*)d_in[9];
  const float* mlp_w2     = (const float*)d_in[10];
  const float* mlp_b2     = (const float*)d_in[11];

  char* ws = (char*)d_ws;
  size_t off = 0;
  auto carve = [&](size_t bytes) -> char* {
    char* p = ws + off;
    off = (off + bytes + 255) & ~(size_t)255;
    return p;
  };
  __bf16* w_qkv  = (__bf16*)carve((size_t)3 * D_ * D_ * 2);       //  6 MB
  __bf16* w_out  = (__bf16*)carve((size_t)D_ * D_ * 2);           //  2 MB
  __bf16* w_m1   = (__bf16*)carve((size_t)MLPH_ * D_ * 2);        //  8 MB
  __bf16* w_m2   = (__bf16*)carve((size_t)D_ * MLPH_ * 2);        //  8 MB
  float*  modp   = (float*) carve((size_t)B_ * SIXD_ * 4);        // 96 KB
  __bf16* xm     = (__bf16*)carve((size_t)NTOK_ * D_ * 2);        // 16 MB (xm / attn_out, serialized reuse)
  float*  qkvbuf = (float*) carve((size_t)3 * B_ * H_ * S_ * DH_ * 4); // 100 MB (qkv f32 / mlp hidden bf16)
  float*  x2     = (float*) carve((size_t)NTOK_ * D_ * 4);        // 32 MB
  __bf16* hbuf   = (__bf16*)qkvbuf;  // qkv dead after attention -> reuse for MLP hidden

  // 1) weights -> bf16 (HBM-bandwidth trivial)
  cvt_bf16_kernel<<<1024, 256, 0, stream>>>(in_proj_w,  w_qkv, 3 * D_ * D_);
  cvt_bf16_kernel<<<1024, 256, 0, stream>>>(out_proj_w, w_out, D_ * D_);
  cvt_bf16_kernel<<<1024, 256, 0, stream>>>(mlp_w1,     w_m1,  MLPH_ * D_);
  cvt_bf16_kernel<<<1024, 256, 0, stream>>>(mlp_w2,     w_m2,  D_ * MLPH_);

  // 2) adaLN modulation vector [B, 6D]
  ada_mod_kernel<<<(B_ * SIXD_ + 255) / 256, 256, 0, stream>>>(c, ada_w, ada_b, modp);

  // 3) LN(x)*(1+sc_msa)+sh_msa -> xm (bf16)
  ln_mod_kernel<<<NTOK_, 256, 0, stream>>>(x, modp, xm, /*sh*/0, /*sc*/D_);

  // 4) qkv = xm @ in_proj_w^T + b  (scatter to [3,B,H,S,64], q pre-scaled)
  gemm_wmma_kernel<<<dim3(64, 24), 256, 0, stream>>>(
      xm, w_qkv, in_proj_b, D_, 3 * D_, MODE_QKV, 0,
      modp, nullptr, nullptr, nullptr, qkvbuf);

  // 5) windowed attention -> attn_out (bf16, reuses xm buffer)
  attn_window_kernel<<<(B_ * H_ * S_) / 8, 256, 0, stream>>>(qkvbuf, xm);

  // 6) x2 = x + g_msa * (attn_out @ out_proj_w^T + b)
  gemm_wmma_kernel<<<dim3(64, 8), 256, 0, stream>>>(
      xm, w_out, out_proj_b, D_, D_, MODE_GATE, /*g_msa*/2 * D_,
      modp, x, x2, nullptr, nullptr);

  // 7) LN(x2)*(1+sc_mlp)+sh_mlp -> xm (bf16)
  ln_mod_kernel<<<NTOK_, 256, 0, stream>>>(x2, modp, xm, /*sh*/3 * D_, /*sc*/4 * D_);

  // 8) h = relu(xm @ mlp_w1^T + b1) -> bf16 (reuses qkv buffer)
  gemm_wmma_kernel<<<dim3(64, 32), 256, 0, stream>>>(
      xm, w_m1, mlp_b1, D_, MLPH_, MODE_RELU, 0,
      modp, nullptr, nullptr, hbuf, nullptr);

  // 9) out = x2 + g_mlp * (h @ mlp_w2^T + b2)
  gemm_wmma_kernel<<<dim3(64, 8), 256, 0, stream>>>(
      hbuf, w_m2, mlp_b2, MLPH_, D_, MODE_GATE, /*g_mlp*/5 * D_,
      modp, x2, (float*)d_out, nullptr, nullptr);
}
